// LLDeepFM_6820408066825
// MI455X (gfx1250) — compile-verified, matrix operand order checked by reading, not run on hardware
//
#include <hip/hip_runtime.h>
#include <hip/hip_bf16.h>

// LLDeepFM on MI455X (gfx1250): anchor-grouped (MoE-style) WMMA MLP.
// B=1024, D=F=39, V=5000, E=8, A=100, K=8, H1=H2=32, C=1000.

typedef __attribute__((ext_vector_type(16))) _Float16 v16h;
typedef __attribute__((ext_vector_type(8)))  _Float16 v8h;
typedef __attribute__((ext_vector_type(8)))  float    v8f;

#define NB   1024   // batch
#define ND   39     // feature dim / fields
#define NV   5000
#define NE   8
#define NA   100
#define NK   8
#define NH   32
#define XPAD 320    // 312 padded to 10 K-chunks of 32
#define XDIM 312
#define NPAIR (NB*NK)
#define NW5  8      // waves per block in k5_mlp

// ---------------------------------------------------------------- init
__global__ __launch_bounds__(128) void k0_init(int* counts) {
    if (threadIdx.x < 128) counts[threadIdx.x] = 0;
}

// ------------------------------------------------- distances + top-8 + softmax
__global__ __launch_bounds__(128) void k1_topk(const float* __restrict__ X,
                                               const float* __restrict__ anchors,
                                               int* __restrict__ idxW,
                                               float* __restrict__ wW,
                                               int* __restrict__ counts,
                                               float* __restrict__ out) {
    const int b = blockIdx.x;
    const int t = threadIdx.x;
    __shared__ float sx[ND];
    __shared__ float dsv[128];
    __shared__ float rv[128];
    __shared__ int   ri[128];
    __shared__ float seld[NK];
    __shared__ int   seli[NK];

    if (t < ND) sx[t] = X[b * ND + t];
    __syncthreads();

    float d = 3.4e38f;
    if (t < NA) {
        float dot = 0.f, aa = 0.f, xx = 0.f;
        #pragma unroll
        for (int j = 0; j < ND; ++j) {
            float av = anchors[t * ND + j];
            float xv = sx[j];
            dot += xv * av; aa += av * av; xx += xv * xv;
        }
        d = xx + aa - 2.f * dot;
    }
    dsv[t] = d;
    __syncthreads();

    for (int k = 0; k < NK; ++k) {
        rv[t] = dsv[t]; ri[t] = t;
        __syncthreads();
        for (int s = 64; s > 0; s >>= 1) {
            if (t < s && rv[t + s] < rv[t]) { rv[t] = rv[t + s]; ri[t] = ri[t + s]; }
            __syncthreads();
        }
        if (t == 0) { seld[k] = rv[0]; seli[k] = ri[0]; dsv[ri[0]] = 3.4e38f; }
        __syncthreads();
    }

    if (t == 0) {
        out[b] = 0.f;  // zero output before k5 atomics (stream-ordered)
        float nd[NK], ex[NK];
        float mx = -3.4e38f;
        #pragma unroll
        for (int k = 0; k < NK; ++k) { nd[k] = -seld[k] * (1.0f / 1000.0f); mx = nd[k] > mx ? nd[k] : mx; }
        float sum = 0.f;
        #pragma unroll
        for (int k = 0; k < NK; ++k) { ex[k] = __expf(nd[k] - mx); sum += ex[k]; }
        float inv = 1.f / sum;
        #pragma unroll
        for (int k = 0; k < NK; ++k) {
            wW[b * NK + k]   = ex[k] * inv;
            idxW[b * NK + k] = seli[k];
            atomicAdd(&counts[seli[k]], 1);
        }
    }
}

// ---------------------------------- embedding gather + FM 2nd order + f16 x
__global__ __launch_bounds__(256) void k2_embed(const int*   __restrict__ Xi,
                                                const float* __restrict__ Xv,
                                                const float* __restrict__ emb,
                                                const float* __restrict__ biasg,
                                                const int*   __restrict__ idxW,
                                                _Float16*    __restrict__ xF,
                                                float*       __restrict__ partial) {
    const int wave = threadIdx.x >> 5;
    const int lane = threadIdx.x & 31;
    const int p = blockIdx.x * 8 + wave;          // grid=1024 -> p < 8192
    const int b = p >> 3;
    const int a = idxW[p];
    float s = 0.f, sq = 0.f;
    _Float16* xp = xF + (size_t)p * XPAD;
    #pragma unroll
    for (int t = 0; t < 10; ++t) {
        const int i = t * 32 + lane;              // i%8 == lane%8 -> fixed e per lane
        float val = 0.f;
        if (i < XDIM) {
            const int f = i >> 3, e = i & 7;
            const int vi = Xi[b * ND + f];
            const float xv = Xv[b * ND + f];
            val = emb[(((size_t)a * ND + f) * NV + vi) * NE + e] * xv;
            s += val; sq += val * val;
        }
        xp[i] = (_Float16)val;                    // zero-pad 312..319 for WMMA
    }
    // sum over fields f: lanes sharing e are {l, l^8, l^16, l^24}
    s  += __shfl_xor(s, 8, 32);  s  += __shfl_xor(s, 16, 32);
    sq += __shfl_xor(sq, 8, 32); sq += __shfl_xor(sq, 16, 32);
    float fm2 = 0.5f * (s * s - sq);
    // sum over the 8 distinct e within each group of 8 lanes
    fm2 += __shfl_xor(fm2, 1, 32);
    fm2 += __shfl_xor(fm2, 2, 32);
    fm2 += __shfl_xor(fm2, 4, 32);
    if (lane == 0) partial[p] = biasg[a] + fm2;
}

// ---------------------------------------------------------------- scan
__global__ void k3_scan(const int* __restrict__ counts,
                        int* __restrict__ offsets,
                        int* __restrict__ cursor) {
    if (threadIdx.x == 0) {
        int acc = 0;
        for (int a = 0; a < NA; ++a) {
            offsets[a] = acc; cursor[a] = acc; acc += counts[a];
        }
    }
}

// -------------------------------------------------------------- scatter
__global__ __launch_bounds__(256) void k4_scatter(const int* __restrict__ idxW,
                                                  int* __restrict__ cursor,
                                                  int* __restrict__ order) {
    const int p = blockIdx.x * 256 + threadIdx.x;
    if (p < NPAIR) {
        const int a = idxW[p];
        const int pos = atomicAdd(&cursor[a], 1);
        order[pos] = p;
    }
}

// ------------------------------------------- per-anchor grouped WMMA MLP
// Fragment K map (16-bit operands, 16x16x32): lanes 0-15 hold K 0..7 (halves
// 0..7) and K 16..23 (halves 8..15); lanes 16-31 hold K 8..15 and 24..31.
__device__ __forceinline__ int kmap(int lane, int h) {
    return ((lane >> 4) << 3) + (h & 7) + ((h >> 3) << 4);
}

__global__ __launch_bounds__(NW5 * 32) void k5_mlp(const float* __restrict__ W1g,
                                                   const float* __restrict__ b1g,
                                                   const float* __restrict__ W2g,
                                                   const float* __restrict__ b2g,
                                                   const _Float16* __restrict__ xF,
                                                   const float* __restrict__ wW,
                                                   const float* __restrict__ partial,
                                                   const int* __restrict__ order,
                                                   const int* __restrict__ counts,
                                                   const int* __restrict__ offsets,
                                                   float* __restrict__ out) {
    const int a    = blockIdx.x;
    const int tid  = threadIdx.x;
    const int lane = tid & 31;
    const int wave = tid >> 5;
    const int cnt  = counts[a];
    if (cnt == 0) return;   // uniform per block

    // W1[a] / W2[a] pre-swizzled into per-lane contiguous B-fragments.
    __shared__ alignas(32) _Float16 sW1[10 * 2 * 32 * 16];  // [kc][nh][lane][16]
    __shared__ alignas(32) _Float16 sW2[2 * 32 * 16];       // [nh][lane][16]
    __shared__ float sb1[NH];
    __shared__ float sb2[NH];
    __shared__ alignas(32) _Float16 sH1[NW5][16 * 32];      // per-wave h1 staging

    for (int u = tid; u < 10 * 2 * 32 * 16; u += NW5 * 32) {
        const int h = u & 15, ln = (u >> 4) & 31, nh = (u >> 9) & 1, kc = u >> 10;
        const int k = kc * 32 + kmap(ln, h);
        const int n = nh * 16 + (ln & 15);
        sW1[u] = (_Float16)((k < XDIM) ? W1g[((size_t)a * XDIM + k) * NH + n] : 0.0f);
    }
    for (int u = tid; u < 2 * 32 * 16; u += NW5 * 32) {
        const int h = u & 15, ln = (u >> 4) & 31, nh = (u >> 9) & 1;
        const int k = kmap(ln, h);
        const int n = nh * 16 + (ln & 15);
        sW2[u] = (_Float16)W2g[((size_t)a * NH + k) * NH + n];
    }
    if (tid < NH) { sb1[tid] = b1g[a * NH + tid]; sb2[tid] = b2g[a * NH + tid]; }
    __syncthreads();

    const int base   = offsets[a];
    const int nTiles = (cnt + 15) >> 4;
    const int m  = lane & 15;
    const int kb = (lane >> 4) << 3;   // 0 or 8
    const float b1v0 = sb1[m], b1v1 = sb1[16 + m];
    const float b2v0 = sb2[m], b2v1 = sb2[16 + m];

    for (int tile = wave; tile < nTiles; tile += NW5) {   // uniform per wave
        const int row = tile * 16 + m;
        const int p = order[base + ((row < cnt) ? row : tile * 16)];
        const _Float16* xp = xF + (size_t)p * XPAD;

        v8f c0, c1;
        #pragma unroll
        for (int r = 0; r < 8; ++r) { c0[r] = b1v0; c1[r] = b1v1; }

        // layer 1: 16 pairs x 32 hidden, K=320 in 10 chunks.
        // Software-pipelined: chunk kc+1's A-fragment loads issue before the
        // WMMAs of chunk kc so L2 latency overlaps matrix issue (B-fragments
        // are re-read from LDS sW1 for every chunk).
        v8h lo = *(const v8h*)(xp + kb);
        v8h hi = *(const v8h*)(xp + kb + 16);
        #pragma unroll
        for (int kc = 0; kc < 10; ++kc) {
            v8h lo_n = {}, hi_n = {};
            if (kc < 9) {
                lo_n = *(const v8h*)(xp + (kc + 1) * 32 + kb);
                hi_n = *(const v8h*)(xp + (kc + 1) * 32 + kb + 16);
            }
            v16h af;
            #pragma unroll
            for (int j = 0; j < 8; ++j) { af[j] = lo[j]; af[8 + j] = hi[j]; }
            const v16h bf0 = *(const v16h*)(sW1 + ((kc * 2 + 0) * 32 + lane) * 16);
            const v16h bf1 = *(const v16h*)(sW1 + ((kc * 2 + 1) * 32 + lane) * 16);
            c0 = __builtin_amdgcn_wmma_f32_16x16x32_f16(false, af, false, bf0, (short)0, c0, false, false);
            c1 = __builtin_amdgcn_wmma_f32_16x16x32_f16(false, af, false, bf1, (short)0, c1, false, false);
            lo = lo_n; hi = hi_n;
        }

        // relu + stage h1 as row-major f16 in per-wave LDS buffer
        _Float16* hp = &sH1[wave][0];
        #pragma unroll
        for (int r = 0; r < 8; ++r) {
            const int mm = r + ((lane >> 4) << 3);
            const float v0 = c0[r] > 0.f ? c0[r] : 0.f;
            const float v1 = c1[r] > 0.f ? c1[r] : 0.f;
            hp[mm * 32 + (lane & 15)]      = (_Float16)v0;
            hp[mm * 32 + 16 + (lane & 15)] = (_Float16)v1;
        }
        // same-wave LDS RAW: DS ops in-order; compiler inserts s_wait_dscnt
        const v8h l2lo = *(const v8h*)(hp + m * 32 + kb);
        const v8h l2hi = *(const v8h*)(hp + m * 32 + kb + 16);
        v16h a2;
        #pragma unroll
        for (int j = 0; j < 8; ++j) { a2[j] = l2lo[j]; a2[8 + j] = l2hi[j]; }
        const v16h w20 = *(const v16h*)(sW2 + (0 * 32 + lane) * 16);
        const v16h w21 = *(const v16h*)(sW2 + (1 * 32 + lane) * 16);
        v8f d0, d1;
        #pragma unroll
        for (int r = 0; r < 8; ++r) { d0[r] = b2v0; d1[r] = b2v1; }
        d0 = __builtin_amdgcn_wmma_f32_16x16x32_f16(false, a2, false, w20, (short)0, d0, false, false);
        d1 = __builtin_amdgcn_wmma_f32_16x16x32_f16(false, a2, false, w21, (short)0, d1, false, false);

        // deep_sum: per-row sum over 32 hidden -> reduce across 16 lanes/group
        float rs[8];
        #pragma unroll
        for (int r = 0; r < 8; ++r) {
            float v = (d0[r] > 0.f ? d0[r] : 0.f) + (d1[r] > 0.f ? d1[r] : 0.f);
            v += __shfl_xor(v, 1, 32);
            v += __shfl_xor(v, 2, 32);
            v += __shfl_xor(v, 4, 32);
            v += __shfl_xor(v, 8, 32);
            rs[r] = v;
        }
        if ((lane & 15) == 0) {     // lanes 0 and 16 -> rows 0..7 / 8..15
            const int half = lane >> 4;
            #pragma unroll
            for (int r = 0; r < 8; ++r) {
                const int grow = tile * 16 + r + (half << 3);
                if (grow < cnt) {
                    const int pp = order[base + grow];
                    const int b  = pp >> 3;
                    const float score = partial[pp] + rs[r];
                    atomicAdd(&out[b], wW[pp] * score);
                }
            }
        }
    }
}

extern "C" void kernel_launch(void* const* d_in, const int* in_sizes, int n_in,
                              void* d_out, int out_size, void* d_ws, size_t ws_size,
                              hipStream_t stream) {
    const float* X       = (const float*)d_in[0];
    const int*   Xi      = (const int*)  d_in[1];
    const float* Xv      = (const float*)d_in[2];
    const float* anchors = (const float*)d_in[3];
    const float* emb     = (const float*)d_in[4];
    const float* biasg   = (const float*)d_in[5];
    const float* W1g     = (const float*)d_in[6];
    const float* b1g     = (const float*)d_in[7];
    const float* W2g     = (const float*)d_in[8];
    const float* b2g     = (const float*)d_in[9];
    float* out = (float*)d_out;

    char* ws = (char*)d_ws;
    int*      idxW    = (int*)     (ws + 0);        // 8192 * 4
    float*    wW      = (float*)   (ws + 32768);    // 8192 * 4
    float*    partial = (float*)   (ws + 65536);    // 8192 * 4
    int*      order   = (int*)     (ws + 98304);    // 8192 * 4
    int*      counts  = (int*)     (ws + 131072);   // 128 * 4
    int*      offsets = (int*)     (ws + 131584);   // 128 * 4
    int*      cursor  = (int*)     (ws + 132096);   // 128 * 4
    _Float16* xF      = (_Float16*)(ws + 132608);   // 8192 * 320 * 2 ≈ 5 MB

    k0_init   <<<1,        128, 0, stream>>>(counts);
    k1_topk   <<<NB,       128, 0, stream>>>(X, anchors, idxW, wW, counts, out);
    k2_embed  <<<NB,       256, 0, stream>>>(Xi, Xv, emb, biasg, idxW, xF, partial);
    k3_scan   <<<1,         32, 0, stream>>>(counts, offsets, cursor);
    k4_scatter<<<32,       256, 0, stream>>>(idxW, cursor, order);
    k5_mlp    <<<NA, NW5 * 32, 0, stream>>>(W1g, b1g, W2g, b2g, xF, wW, partial,
                                            order, counts, offsets, out);
}
